// BasicBlock_20160576487544
// MI455X (gfx1250) — matrix operand order repaired and checked by reading, not compiled, not run on hardware
//
#include <hip/hip_runtime.h>

// ---------------------------------------------------------------------------
// BasicBlock (sparse conv -> BN -> ReLU -> sparse conv -> BN -> +res -> ReLU)
// MI455X / gfx1250: wave32, v_wmma_f32_16x16x32_bf16, f32 accumulation,
// async global->LDS double-buffered weight tiles (ASYNCcnt pipeline).
// ---------------------------------------------------------------------------

typedef __attribute__((ext_vector_type(4)))  __bf16 v4bf;
typedef __attribute__((ext_vector_type(8)))  __bf16 v8bf;
typedef __attribute__((ext_vector_type(16))) __bf16 v16bf;
typedef __attribute__((ext_vector_type(8)))  float  v8f;

#define C_CH   128
#define K_OFF  27
#define MT     128          // rows per block tile
#define LDS_LD 136          // padded LDS row stride (elements), 16B aligned
#define BN_EPS 1e-5f

#if defined(__HIP_DEVICE_COMPILE__) && __has_builtin(__builtin_amdgcn_s_wait_asynccnt)
#define WAIT_ASYNC(n) __builtin_amdgcn_s_wait_asynccnt(n)
#else
#define WAIT_ASYNC(n) asm volatile("s_wait_asynccnt %0" ::"n"(n) : "memory")
#endif

// One per-lane 16-byte async copy: global -> LDS (gfx1250 GLOBAL_LOAD_ASYNC_TO_LDS_B128).
__device__ __forceinline__ void async_copy16(unsigned int lds_addr, const void* gptr)
{
    asm volatile("global_load_async_to_lds_b128 %0, %1, off"
                 :: "v"(lds_addr), "v"((unsigned long long)(uintptr_t)gptr)
                 : "memory");
}

// ---------------------------------------------------------------------------
// One-time weight prep: W[k][cin][cout] f32  ->  Wt[k][cout][cin] bf16.
// Keeps the hot loop free of weight transpose/convert work; bf16 tensor
// (884 KB) stays resident in the 192 MB L2 across all 1563 tiles.
// ---------------------------------------------------------------------------
__global__ __launch_bounds__(256)
void prep_weights_kernel(const float* __restrict__ W, __bf16* __restrict__ Wt)
{
    const float* wk  = W  + (size_t)blockIdx.x * (C_CH * C_CH);
    __bf16*      wtk = Wt + (size_t)blockIdx.x * (C_CH * C_CH);
    for (int q = threadIdx.x; q < (C_CH * C_CH) / 4; q += 256) {
        float4 w4 = ((const float4*)wk)[q];
        const int cin   = q >> 5;           // (q*4)/128
        const int cout0 = (q & 31) * 4;
        wtk[(cout0 + 0) * C_CH + cin] = (__bf16)w4.x;
        wtk[(cout0 + 1) * C_CH + cin] = (__bf16)w4.y;
        wtk[(cout0 + 2) * C_CH + cin] = (__bf16)w4.z;
        wtk[(cout0 + 3) * C_CH + cin] = (__bf16)w4.w;
    }
}

// ---------------------------------------------------------------------------
// Fused sparse-conv tile kernel.
//   out[r, :] = sum_k mask[r,k] * in[idx[r,k], :] @ W[k]      (128x128 GEMMs)
// Also accumulates per-channel sum / sum-of-squares for batch norm.
// ---------------------------------------------------------------------------
template <bool IN_BF16>
__global__ __launch_bounds__(256)
void sparse_conv_wmma(const void* __restrict__ in,
                      const __bf16* __restrict__ Wt,       // [27,128,128] bf16, [k][cout][cin]
                      const long long* __restrict__ idx,   // [N,27] int64
                      const int* __restrict__ msk,         // [N,27] int32
                      float* __restrict__ out,             // [N,128] f32
                      float* __restrict__ bn_sum,          // [128]
                      float* __restrict__ bn_sq,           // [128]
                      int n_rows)
{
    __shared__ __bf16 As[MT * LDS_LD];           // gathered activations (bf16)
    __shared__ __bf16 Ws[2][C_CH * LDS_LD];      // double-buffered Wt[k] tile
    __shared__ float  s_sum[C_CH];
    __shared__ float  s_sq[C_CH];

    const int tid  = threadIdx.x;
    const int wv   = tid >> 5;                   // wave id 0..7
    const int lane = tid & 31;                   // wave32 lane
    const int row0 = blockIdx.x * MT;

    if (tid < C_CH) { s_sum[tid] = 0.0f; s_sq[tid] = 0.0f; }

    v8f acc[8];
#pragma unroll
    for (int n = 0; n < 8; ++n) acc[n] = (v8f)0.0f;

    // Issue weight tile k: 2048 x 16B chunks, 8 per thread, scattered into the
    // padded LDS layout via per-lane LDS destination addresses.
    auto issue_weights = [&](int buf, int k) {
        const __bf16* wt = Wt + (size_t)k * (C_CH * C_CH);
#pragma unroll
        for (int i = 0; i < 8; ++i) {
            const int q    = tid + i * 256;      // 16B chunk index
            const int cout = q >> 4;
            const int c16  = q & 15;
            async_copy16((unsigned int)(uintptr_t)&Ws[buf][cout * LDS_LD + c16 * 8],
                         wt + q * 8);
        }
    };

    issue_weights(0, 0);                         // prologue: stage 0 in flight

    for (int k = 0; k < K_OFF; ++k) {
        // ---- gather A tile: 2 threads per row, 64 channels each ----
        {
            const int r    = tid >> 1;
            const int half = (tid & 1) * 64;
            const int g    = row0 + r;
            long long nb = 0;
            int m = 0;
            if (g < n_rows) {
                nb = idx[(long long)g * K_OFF + k];
                m  = msk[(long long)g * K_OFF + k];
            }
            __bf16* dst = &As[r * LDS_LD + half];
            if (m != 0) {
                if constexpr (IN_BF16) {
                    const uint4* src = (const uint4*)((const __bf16*)in + nb * C_CH + half);
                    uint4* d4 = (uint4*)dst;
#pragma unroll
                    for (int i = 0; i < 8; ++i) d4[i] = src[i];   // 64 bf16
                } else {
                    const float4* src = (const float4*)((const float*)in + nb * C_CH + half);
#pragma unroll
                    for (int i = 0; i < 16; ++i) {
                        float4 v = src[i];
                        v4bf b;
                        b.x = (__bf16)v.x; b.y = (__bf16)v.y;
                        b.z = (__bf16)v.z; b.w = (__bf16)v.w;
                        *(v4bf*)(dst + i * 4) = b;
                    }
                }
            } else {
                uint4 z = {0u, 0u, 0u, 0u};
                uint4* d4 = (uint4*)dst;
#pragma unroll
                for (int i = 0; i < 8; ++i) d4[i] = z;            // 128 bytes
            }
        }

        // ---- pipeline weights: issue k+1, wait for k (8 in flight allowed) ----
        if (k + 1 < K_OFF) {
            issue_weights((k + 1) & 1, k + 1);
            WAIT_ASYNC(8);                        // stage k landed, k+1 in flight
        } else {
            WAIT_ASYNC(0);
        }
        __syncthreads();

        // ---- WMMA: wave wv owns rows [wv*16, wv*16+16), all 8 col tiles ----
        const __bf16* Wb = Ws[k & 1];
        const int arow = (wv * 16 + (lane & 15)) * LDS_LD;
        const int akb  = (lane < 16) ? 0 : 8;     // A: 16-bit K split per ISA
        const int bkb  = (lane < 16) ? 0 : 8;     // B: 16-bit K split per ISA
#pragma unroll
        for (int kk = 0; kk < C_CH; kk += 32) {
            v8bf a0 = *(const v8bf*)&As[arow + kk + akb];
            v8bf a1 = *(const v8bf*)&As[arow + kk + akb + 16];
            v16bf a = __builtin_shufflevector(a0, a1, 0, 1, 2, 3, 4, 5, 6, 7,
                                              8, 9, 10, 11, 12, 13, 14, 15);
#pragma unroll
            for (int n = 0; n < 8; ++n) {
                const int bcol = (n * 16 + (lane & 15)) * LDS_LD;
                v8bf b0 = *(const v8bf*)&Wb[bcol + kk + bkb];
                v8bf b1 = *(const v8bf*)&Wb[bcol + kk + bkb + 16];
                v16bf b = __builtin_shufflevector(b0, b1, 0, 1, 2, 3, 4, 5, 6, 7,
                                                  8, 9, 10, 11, 12, 13, 14, 15);
                acc[n] = __builtin_amdgcn_wmma_f32_16x16x32_bf16(
                    false, a, false, b, (short)0, acc[n], false, false);
            }
        }

        __syncthreads();   // before next k overwrites As
    }

    // ---- epilogue: store f32 result + fused BN partial statistics ----
    const int colbase = lane & 15;
    const int rbase   = row0 + wv * 16 + ((lane < 16) ? 0 : 8);
#pragma unroll
    for (int n = 0; n < 8; ++n) {
        const int col = n * 16 + colbase;
        float ps = 0.0f, pq = 0.0f;
#pragma unroll
        for (int j = 0; j < 8; ++j) {
            const int grow = rbase + j;
            const float v = acc[n][j];
            if (grow < n_rows) {
                out[(long long)grow * C_CH + col] = v;
                ps += v;
                pq += v * v;
            }
        }
        atomicAdd(&s_sum[col], ps);   // ds_add_f32
        atomicAdd(&s_sq[col], pq);
    }
    __syncthreads();
    if (tid < C_CH) {
        atomicAdd(&bn_sum[tid], s_sum[tid]);  // global_atomic_add_f32
        atomicAdd(&bn_sq[tid],  s_sq[tid]);
    }
}

// ---------------------------------------------------------------------------
__global__ void zero_stats_kernel(float* p)
{
    p[blockIdx.x * 256 + threadIdx.x] = 0.0f;
}

__global__ void bn_finalize_kernel(const float* __restrict__ sum,
                                   const float* __restrict__ sq,
                                   const float* __restrict__ gamma,
                                   const float* __restrict__ beta,
                                   float* __restrict__ scale,
                                   float* __restrict__ shift,
                                   int n_rows)
{
    const int c = threadIdx.x;
    if (c < C_CH) {
        const float inv_n = 1.0f / (float)n_rows;
        const float mu  = sum[c] * inv_n;
        const float var = sq[c] * inv_n - mu * mu;
        const float sc  = gamma[c] * rsqrtf(var + BN_EPS);
        scale[c] = sc;
        shift[c] = beta[c] - mu * sc;
    }
}

// BN + ReLU, requantize to bf16 for the second conv's gathers.
__global__ __launch_bounds__(256)
void bn_relu_to_bf16_kernel(const float* __restrict__ in,
                            const float* __restrict__ scale,
                            const float* __restrict__ shift,
                            __bf16* __restrict__ out,
                            long long total)
{
    const long long i = ((long long)blockIdx.x * blockDim.x + threadIdx.x) * 4;
    if (i >= total) return;
    const float4 v = *(const float4*)(in + i);
    const int c = (int)(i & (C_CH - 1));
    v4bf b;
    b.x = (__bf16)fmaxf(v.x * scale[c + 0] + shift[c + 0], 0.0f);
    b.y = (__bf16)fmaxf(v.y * scale[c + 1] + shift[c + 1], 0.0f);
    b.z = (__bf16)fmaxf(v.z * scale[c + 2] + shift[c + 2], 0.0f);
    b.w = (__bf16)fmaxf(v.w * scale[c + 3] + shift[c + 3], 0.0f);
    *(v4bf*)(out + i) = b;
}

// BN + residual + ReLU -> final f32 output.
__global__ __launch_bounds__(256)
void bn_add_relu_kernel(const float* __restrict__ conv,
                        const float* __restrict__ res,
                        const float* __restrict__ scale,
                        const float* __restrict__ shift,
                        float* __restrict__ out,
                        long long total)
{
    const long long i = ((long long)blockIdx.x * blockDim.x + threadIdx.x) * 4;
    if (i >= total) return;
    const float4 v = *(const float4*)(conv + i);
    const float4 r = *(const float4*)(res + i);
    const int c = (int)(i & (C_CH - 1));
    float4 o;
    o.x = fmaxf(v.x * scale[c + 0] + shift[c + 0] + r.x, 0.0f);
    o.y = fmaxf(v.y * scale[c + 1] + shift[c + 1] + r.y, 0.0f);
    o.z = fmaxf(v.z * scale[c + 2] + shift[c + 2] + r.z, 0.0f);
    o.w = fmaxf(v.w * scale[c + 3] + shift[c + 3] + r.w, 0.0f);
    *(float4*)(out + i) = o;
}

// ---------------------------------------------------------------------------
extern "C" void kernel_launch(void* const* d_in, const int* in_sizes, int n_in,
                              void* d_out, int out_size, void* d_ws, size_t ws_size,
                              hipStream_t stream)
{
    const float*     x   = (const float*)d_in[0];
    const float*     W1  = (const float*)d_in[1];
    const float*     g1  = (const float*)d_in[2];
    const float*     b1  = (const float*)d_in[3];
    const float*     W2  = (const float*)d_in[4];
    const float*     g2  = (const float*)d_in[5];
    const float*     b2  = (const float*)d_in[6];
    const long long* idx = (const long long*)d_in[7];   // int64 indices
    const int*       msk = (const int*)d_in[8];

    const int n_rows = in_sizes[0] / C_CH;
    const long long total = (long long)n_rows * C_CH;
    const size_t wbytes = (size_t)K_OFF * C_CH * C_CH * sizeof(__bf16);  // 884,736

    // workspace layout:
    //   [0, 4096)             : BN stats + scale/shift (1024 f32)
    //   [4096, +wbytes)       : Wt1 bf16 transposed
    //   [.., +wbytes)         : Wt2 bf16 transposed
    //   [.., +total*4)        : f32 conv output (reused conv1/conv2)
    //   [.., +total*2)        : bf16 hidden activations
    char*  ws     = (char*)d_ws;
    float* stats  = (float*)ws;
    float* sum1   = stats +   0, *sq1 = stats + 128;
    float* sum2   = stats + 256, *sq2 = stats + 384;
    float* scale1 = stats + 512, *shift1 = stats + 640;
    float* scale2 = stats + 768, *shift2 = stats + 896;
    __bf16* Wt1   = (__bf16*)(ws + 4096);
    __bf16* Wt2   = (__bf16*)(ws + 4096 + wbytes);
    float*  outbuf = (float*)(ws + 4096 + 2 * wbytes);
    __bf16* hbuf   = (__bf16*)(ws + 4096 + 2 * wbytes + (size_t)total * sizeof(float));

    zero_stats_kernel<<<2, 256, 0, stream>>>(stats);   // clears sums 0..511
    prep_weights_kernel<<<K_OFF, 256, 0, stream>>>(W1, Wt1);
    prep_weights_kernel<<<K_OFF, 256, 0, stream>>>(W2, Wt2);

    const int nblk = (n_rows + MT - 1) / MT;
    const int eblk = (int)((total / 4 + 255) / 256);

    // conv1 (f32 input) + BN1 stats
    sparse_conv_wmma<false><<<nblk, 256, 0, stream>>>(
        (const void*)x, Wt1, idx, msk, outbuf, sum1, sq1, n_rows);
    bn_finalize_kernel<<<1, 128, 0, stream>>>(sum1, sq1, g1, b1, scale1, shift1, n_rows);
    bn_relu_to_bf16_kernel<<<eblk, 256, 0, stream>>>(outbuf, scale1, shift1, hbuf, total);

    // conv2 (bf16 input) + BN2 stats
    sparse_conv_wmma<true><<<nblk, 256, 0, stream>>>(
        (const void*)hbuf, Wt2, idx, msk, outbuf, sum2, sq2, n_rows);
    bn_finalize_kernel<<<1, 128, 0, stream>>>(sum2, sq2, g2, b2, scale2, shift2, n_rows);

    // BN2 + identity residual + ReLU -> d_out
    bn_add_relu_kernel<<<eblk, 256, 0, stream>>>(outbuf, x, scale2, shift2,
                                                 (float*)d_out, total);
}
